// SparseMixtureOfExperts_73177652789484
// MI455X (gfx1250) — matrix operand order repaired and checked by reading, not compiled, not run on hardware
//
#include <hip/hip_runtime.h>
#include <hip/hip_bf16.h>
#include <math.h>

// ---------------------------------------------------------------------------
// MoE fused kernel for gfx1250 (MI455X): all matmuls via v_wmma_f32_16x16x32_f16
// Roofline: 146 GFLOP vs 128MB HBM traffic -> compute bound -> matrix cores.
// GELU: branch-free tanh-form (v_exp_f32 + v_rcp_f32), vectorizes to v_pk_* f32.
// Expert weights are software-pipelined through LDS in half-matrix stages so
// the L2->LDS copy of stage k+1 overlaps the WMMA compute of stage k.
// ---------------------------------------------------------------------------

typedef _Float16 v16h __attribute__((ext_vector_type(16)));
typedef _Float16 v8h  __attribute__((ext_vector_type(8)));
typedef float    v8f  __attribute__((ext_vector_type(8)));

#define B_ROWS  131072
#define D_IN    128
#define D_OUT   128
#define NE      8
#define H_EXP   256
#define H_GATE  128

#define MB       128     // rows per workgroup
#define NWAVES   8       // 8 waves x 16 rows
#define NTHREADS 256

// padded LDS strides (in halves) to rotate banks for 16-byte strided reads
#define LDX  136
#define LDG  136
#define LDH  264
#define LDE1 136
#define LDE2 264

// workspace (f16) layout, offsets in halves
#define WS_GW1T 0
#define WS_GW2T (WS_GW1T + 128*128)
#define WS_GW3T (WS_GW2T + 128*128)
#define WS_EW1T (WS_GW3T + 8*128)
#define WS_EW2T (WS_EW1T + 8*256*128)
#define WS_TOTAL (WS_EW2T + 8*128*256)     // 558080 halves = ~1.07 MB

// dynamic LDS layout (bytes): [x f16][softmax w f32][overlaid region f16]
#define LDS_X_HALVES (128*LDX)             // 17408
#define LDS_W_OFF    (LDS_X_HALVES*2)      // 34816
#define LDS_R_OFF    (LDS_W_OFF + 128*8*4) // 38912
#define R_EW1_OFF    (NWAVES*16*LDH)       // h scratch first: 33792 halves
#define R_EW2_OFF    (R_EW1_OFF + 256*LDE1)// 68608
#define R_GW1_OFF    R_EW1_OFF             // gate phase overlays expert region
#define R_GW2_OFF    (R_GW1_OFF + 128*LDG)
#define R_GW3_OFF    (R_GW2_OFF + 128*LDG)
#define R_HALVES     (R_EW2_OFF + 128*LDE2)// 102400
#define LDS_BYTES    (LDS_R_OFF + R_HALVES*2) // 243712 (< 320KB WGP LDS)

// Branch-free tanh-form GELU:
//   gelu(x) = 0.5*x*(1 + tanh(0.79788456*(x + 0.044715*x^3)))
//   tanh(z) = 1 - 2/(exp(2z)+1)  -> one v_exp_f32 + one v_rcp_f32, no branches.
__device__ __forceinline__ float gelu_fast(float x) {
    float x3 = x * x * x;
    float u  = 1.5957691216f * x + 0.0713548162726f * x3;   // 2*z
    float t  = 1.0f - 2.0f * __builtin_amdgcn_rcpf(__expf(u) + 1.0f);
    return 0.5f * x * (1.0f + t);
}

// A fragment (16x32 f16): p = &base[m*ld + k0 + kg*16]
__device__ __forceinline__ v16h frag_a(const _Float16* p) {
    union { v16h v; v8h h[2]; } u;
    u.h[0] = *(const v8h*)(p);
    u.h[1] = *(const v8h*)(p + 8);
    return u.v;
}
// B fragment (32x16 f16) from transposed weights W^T[N][K]:
// p = &bT[(n0+n)*ld + k0 + kg*16]
__device__ __forceinline__ v16h frag_b(const _Float16* p) {
    union { v16h v; v8h h[2]; } u;
    u.h[0] = *(const v8h*)(p);
    u.h[1] = *(const v8h*)(p + 8);
    return u.v;
}

#define WMMA_F32_F16(a, b, c) \
    __builtin_amdgcn_wmma_f32_16x16x32_f16(false, (a), false, (b), (short)0, (c), false, false)

// ---------------------------------------------------------------------------
// Prep: convert all weights f32 -> f16, transposed to [N][K] row-major in ws.
// ---------------------------------------------------------------------------
__global__ void moe_prep_weights(const float* __restrict__ gw1,
                                 const float* __restrict__ gw2,
                                 const float* __restrict__ gw3,
                                 const float* __restrict__ ew1,
                                 const float* __restrict__ ew2,
                                 _Float16* __restrict__ ws) {
    int i = blockIdx.x * blockDim.x + threadIdx.x;
    if (i < 128*128) {                       // gw1t[n][k] = gw1[k][n]
        int n = i >> 7, k = i & 127;
        ws[WS_GW1T + i] = (_Float16)gw1[k*128 + n];
        return;
    }
    i -= 128*128;
    if (i < 128*128) {                       // gw2t
        int n = i >> 7, k = i & 127;
        ws[WS_GW2T + i] = (_Float16)gw2[k*128 + n];
        return;
    }
    i -= 128*128;
    if (i < 8*128) {                         // gw3t[e][k] = gw3[k][e]
        int n = i >> 7, k = i & 127;
        ws[WS_GW3T + i] = (_Float16)gw3[k*8 + n];
        return;
    }
    i -= 8*128;
    if (i < 8*256*128) {                     // ew1t[e][h][k] = ew1[e][k][h]
        int e = i >> 15, r = i & 32767;
        int h = r >> 7, k = r & 127;
        ws[WS_EW1T + i] = (_Float16)ew1[e*32768 + k*256 + h];
        return;
    }
    i -= 8*256*128;
    if (i < 8*128*256) {                     // ew2t[e][o][h] = ew2[e][h][o]
        int e = i >> 15, r = i & 32767;
        int o = r >> 8, hh = r & 255;
        ws[WS_EW2T + i] = (_Float16)ew2[e*32768 + hh*128 + o];
        return;
    }
}

// cooperative f16 global->LDS copy with padded destination stride
__device__ __forceinline__ void copy16(_Float16* __restrict__ dst,
                                       const _Float16* __restrict__ src,
                                       int rows, int cols, int ldd) {
    const int cpr = cols >> 3;               // 8-half (16B) chunks per row
    const int total = rows * cpr;
    for (int i = threadIdx.x; i < total; i += NTHREADS) {
        int r = i / cpr, c = i - r * cpr;
        *(v8h*)(dst + r*ldd + c*8) = *(const v8h*)(src + i*8);
    }
}

// ---------------------------------------------------------------------------
// Fused MoE: gate MLP + softmax + 8 experts, all WMMA, one HBM pass.
// ---------------------------------------------------------------------------
__global__ void __launch_bounds__(NTHREADS)
moe_fused_kernel(const float* __restrict__ x,
                 const float* __restrict__ gb1,
                 const float* __restrict__ gb2,
                 const float* __restrict__ gb3,
                 const float* __restrict__ eb1,
                 const float* __restrict__ eb2,
                 const _Float16* __restrict__ ws,
                 float* __restrict__ out) {
    extern __shared__ char smem[];
    _Float16* xlds   = (_Float16*)smem;
    float*    wsoft  = (float*)(smem + LDS_W_OFF);
    _Float16* region = (_Float16*)(smem + LDS_R_OFF);
    _Float16* hlds   = region;
    // expert weight buffers, split in halves for the 4-stage pipeline
    _Float16* ew1A = region + R_EW1_OFF;            // 128 rows x LDE1 (n-tiles 0..7)
    _Float16* ew1B = ew1A + 128*LDE1;               // 128 rows x LDE1 (n-tiles 8..15)
    _Float16* ew2A = region + R_EW2_OFF;            // 64 rows x LDE2 (n-tiles 0..3)
    _Float16* ew2B = ew2A + 64*LDE2;                // 64 rows x LDE2 (n-tiles 4..7)
    // gate-phase overlays (gate finishes before expert staging begins)
    _Float16* gw1lds = region + R_GW1_OFF;
    _Float16* gw2lds = region + R_GW2_OFF;
    _Float16* gw3lds = region + R_GW3_OFF;

    const int row0 = blockIdx.x * MB;
    const int wv   = threadIdx.x >> 5;       // wave id (wave32)
    const int lane = threadIdx.x & 31;
    const int lm   = lane & 15;              // A: row m   | B/C: column n
    const int lg   = lane >> 4;              // A: k-group | C: m-group

    // ---- stage x tile [128][128] f32 -> f16 LDS -------------------------
    {
        const float4* xs = (const float4*)(x + (size_t)row0 * D_IN);
        const int total = MB * (D_IN / 4);
        for (int i = threadIdx.x; i < total; i += NTHREADS) {
            int r = i >> 5, c = i & 31;
            float4 f = xs[i];
            _Float16* d = xlds + r*LDX + c*4;
            d[0] = (_Float16)f.x; d[1] = (_Float16)f.y;
            d[2] = (_Float16)f.z; d[3] = (_Float16)f.w;
        }
    }
    // ---- stage gate weights (transposed f16 from ws) --------------------
    copy16(gw1lds, ws + WS_GW1T, 128, 128, LDG);
    copy16(gw2lds, ws + WS_GW2T, 128, 128, LDG);
    copy16(gw3lds, ws + WS_GW3T,   8, 128, LDG);
    __syncthreads();

    _Float16* hme = hlds + wv * 16 * LDH;    // per-wave [16][LDH] scratch

    // x A-fragments are reused by gate layer 1 and every expert layer 1
    v16h ax[4];
#pragma unroll
    for (int kt = 0; kt < 4; ++kt)
        ax[kt] = frag_a(xlds + lm*LDX + kt*32 + lg*16);

    // ---- gate layer 1: g1 = gelu(x @ gw1 + gb1) -------------------------
#pragma unroll
    for (int nt = 0; nt < 8; ++nt) {
        v8f c = {};
#pragma unroll
        for (int kt = 0; kt < 4; ++kt)
            c = WMMA_F32_F16(ax[kt], frag_b(gw1lds + (nt*16 + lm)*LDG + kt*32 + lg*16), c);
        float bias = gb1[nt*16 + lm];
#pragma unroll
        for (int r = 0; r < 8; ++r)
            hme[(lg*8 + r)*LDH + nt*16 + lm] = (_Float16)gelu_fast(c[r] + bias);
    }
    // ---- gate layer 2 (read g1 frags first, then overwrite with g2) ----
    v16h ag[4];
#pragma unroll
    for (int kt = 0; kt < 4; ++kt)
        ag[kt] = frag_a(hme + lm*LDH + kt*32 + lg*16);
#pragma unroll
    for (int nt = 0; nt < 8; ++nt) {
        v8f c = {};
#pragma unroll
        for (int kt = 0; kt < 4; ++kt)
            c = WMMA_F32_F16(ag[kt], frag_b(gw2lds + (nt*16 + lm)*LDG + kt*32 + lg*16), c);
        float bias = gb2[nt*16 + lm];
#pragma unroll
        for (int r = 0; r < 8; ++r)
            hme[(lg*8 + r)*LDH + nt*16 + lm] = (_Float16)gelu_fast(c[r] + bias);
    }
    // ---- gate layer 3 (N=8) + softmax -----------------------------------
#pragma unroll
    for (int kt = 0; kt < 4; ++kt)
        ag[kt] = frag_a(hme + lm*LDH + kt*32 + lg*16);
    {
        v8f c = {};
#pragma unroll
        for (int kt = 0; kt < 4; ++kt)
            c = WMMA_F32_F16(ag[kt], frag_b(gw3lds + lm*LDG + kt*32 + lg*16), c);
        if (lm < 8) {
            float bias = gb3[lm];
#pragma unroll
            for (int r = 0; r < 8; ++r)
                wsoft[(wv*16 + lg*8 + r)*8 + lm] = gelu_fast(c[r] + bias);
        }
    }
    if (lane < 16) {                         // per-row softmax over 8 experts
        float* wr = wsoft + (wv*16 + lane)*8;
        float mx = wr[0];
#pragma unroll
        for (int e = 1; e < NE; ++e) mx = fmaxf(mx, wr[e]);
        float ex[NE]; float s = 0.f;
#pragma unroll
        for (int e = 0; e < NE; ++e) { ex[e] = __expf(wr[e] - mx); s += ex[e]; }
        float inv = __builtin_amdgcn_rcpf(s);
#pragma unroll
        for (int e = 0; e < NE; ++e) wr[e] = ex[e] * inv;
    }

    // ---- experts: 4-stage pipeline, copy of next half overlaps compute --
    v8f acc[8];
#pragma unroll
    for (int nt = 0; nt < 8; ++nt) acc[nt] = (v8f){};

    __syncthreads();                               // gate reads of overlay done
    copy16(ew1A, ws + WS_EW1T, 128, 128, LDE1);    // prologue: ew1A[0]
    __syncthreads();

    for (int e = 0; e < NE; ++e) {
        const float* eb1e = eb1 + e*H_EXP;
        const float* eb2e = eb2 + e*D_OUT;
        const _Float16* w1src = ws + WS_EW1T + e*H_EXP*D_IN;
        const _Float16* w2src = ws + WS_EW2T + e*D_OUT*H_EXP;

        // stage 1: L1 n-tiles 0..7 from ew1A  ||  copy ew1B[e]
#pragma unroll
        for (int nt = 0; nt < 8; ++nt) {
            v8f c = {};
#pragma unroll
            for (int kt = 0; kt < 4; ++kt)
                c = WMMA_F32_F16(ax[kt], frag_b(ew1A + (nt*16 + lm)*LDE1 + kt*32 + lg*16), c);
            float bias = eb1e[nt*16 + lm];
#pragma unroll
            for (int r = 0; r < 8; ++r)
                hme[(lg*8 + r)*LDH + nt*16 + lm] = (_Float16)gelu_fast(c[r] + bias);
        }
        copy16(ew1B, w1src + 128*128, 128, 128, LDE1);
        if (e + 1 < NE)                            // warm L2 (global_prefetch_b8)
            __builtin_prefetch(w1src + H_EXP*D_IN + (threadIdx.x << 7), 0, 0);
        __syncthreads();

        // stage 2: L1 n-tiles 8..15 from ew1B  ||  copy ew2A[e]
#pragma unroll
        for (int nt = 0; nt < 8; ++nt) {
            v8f c = {};
#pragma unroll
            for (int kt = 0; kt < 4; ++kt)
                c = WMMA_F32_F16(ax[kt], frag_b(ew1B + (nt*16 + lm)*LDE1 + kt*32 + lg*16), c);
            float bias = eb1e[(8 + nt)*16 + lm];
#pragma unroll
            for (int r = 0; r < 8; ++r)
                hme[(lg*8 + r)*LDH + (8 + nt)*16 + lm] = (_Float16)gelu_fast(c[r] + bias);
        }
        copy16(ew2A, w2src, 64, 256, LDE2);
        __syncthreads();

        // h A-fragments (own wave's h, all K=256)
        v16h ah[8];
#pragma unroll
        for (int kt = 0; kt < 8; ++kt)
            ah[kt] = frag_a(hme + lm*LDH + kt*32 + lg*16);
        float wexp[8];
#pragma unroll
        for (int r = 0; r < 8; ++r)
            wexp[r] = wsoft[(wv*16 + lg*8 + r)*8 + e];

        // stage 3: L2 n-tiles 0..3 from ew2A  ||  copy ew2B[e]
#pragma unroll
        for (int nt = 0; nt < 4; ++nt) {
            v8f c = {};
#pragma unroll
            for (int kt = 0; kt < 8; ++kt)
                c = WMMA_F32_F16(ah[kt], frag_b(ew2A + (nt*16 + lm)*LDE2 + kt*32 + lg*16), c);
            float bias = eb2e[nt*16 + lm];
#pragma unroll
            for (int r = 0; r < 8; ++r)
                acc[nt][r] += wexp[r] * gelu_fast(c[r] + bias);
        }
        copy16(ew2B, w2src + 64*256, 64, 256, LDE2);
        __syncthreads();

        // stage 4: L2 n-tiles 4..7 from ew2B  ||  copy ew1A[e+1]
#pragma unroll
        for (int nt = 0; nt < 4; ++nt) {
            v8f c = {};
#pragma unroll
            for (int kt = 0; kt < 8; ++kt)
                c = WMMA_F32_F16(ah[kt], frag_b(ew2B + (nt*16 + lm)*LDE2 + kt*32 + lg*16), c);
            float bias = eb2e[(4 + nt)*16 + lm];
#pragma unroll
            for (int r = 0; r < 8; ++r)
                acc[4 + nt][r] += wexp[r] * gelu_fast(c[r] + bias);
        }
        if (e + 1 < NE)
            copy16(ew1A, ws + WS_EW1T + (e + 1)*H_EXP*D_IN, 128, 128, LDE1);
        __syncthreads();
    }

    // ---- write out [16][128] f32 ----------------------------------------
    float* orow = out + (size_t)(row0 + wv*16) * D_OUT;
#pragma unroll
    for (int nt = 0; nt < 8; ++nt)
#pragma unroll
        for (int r = 0; r < 8; ++r)
            orow[(lg*8 + r)*D_OUT + nt*16 + lm] = acc[nt][r];
}

// ---------------------------------------------------------------------------
extern "C" void kernel_launch(void* const* d_in, const int* in_sizes, int n_in,
                              void* d_out, int out_size, void* d_ws, size_t ws_size,
                              hipStream_t stream) {
    const float* x   = (const float*)d_in[0];
    const float* gw1 = (const float*)d_in[1];
    const float* gb1 = (const float*)d_in[2];
    const float* gw2 = (const float*)d_in[3];
    const float* gb2 = (const float*)d_in[4];
    const float* gw3 = (const float*)d_in[5];
    const float* gb3 = (const float*)d_in[6];
    const float* ew1 = (const float*)d_in[7];
    const float* eb1 = (const float*)d_in[8];
    const float* ew2 = (const float*)d_in[9];
    const float* eb2 = (const float*)d_in[10];
    float* out = (float*)d_out;
    _Float16* ws = (_Float16*)d_ws;

    // 1) convert + transpose all weights to f16 in workspace (stays L2-resident)
    {
        int total = WS_TOTAL;
        int grid = (total + NTHREADS - 1) / NTHREADS;
        moe_prep_weights<<<grid, NTHREADS, 0, stream>>>(gw1, gw2, gw3, ew1, ew2, ws);
    }
    // 2) fused gate + experts, 128 rows per workgroup, WMMA everywhere
    {
        int grid = B_ROWS / MB;   // 1024
        moe_fused_kernel<<<grid, NTHREADS, LDS_BYTES, stream>>>(
            x, gb1, gb2, gb3, eb1, eb2, ws, out);
    }
}
